// KnnModule_41566693490872
// MI455X (gfx1250) — compile-verified, matrix operand order checked by reading, not run on hardware
//
#include <hip/hip_runtime.h>
#include <hip/hip_bf16.h>
#include <math.h>

// ---------------------------------------------------------------------------
// KNN classifier for MI455X (gfx1250, wave32):
//   sim = normalize(F) @ normalize(T)^T   via v_wmma_f32_16x16x32_bf16
//   per-row top-20, softmax(/0.07), scatter votes for k=10 and k=20.
// ---------------------------------------------------------------------------

typedef __attribute__((ext_vector_type(16))) __bf16 v16bf;
typedef __attribute__((ext_vector_type(8)))  __bf16 v8bf;
typedef __attribute__((ext_vector_type(8)))  float  v8f;

#define NCHUNK      512          // train columns per GEMM block
#define LDS_STRIDE  516          // 512 + 4 pad (breaks LDS bank aliasing)
#define KTOP        20
#define KSMALL      10
#define TINV        (1.0f / 0.07f)
#define NUM_CLASSES 1000

// sched_group_barrier masks
#define SGB_WMMA     0x008
#define SGB_VMEMREAD 0x020

__device__ __forceinline__ unsigned short f32_to_bf16_rne(float f) {
    unsigned int u = __builtin_bit_cast(unsigned int, f);
    u += 0x7FFFu + ((u >> 16) & 1u);   // round-to-nearest-even
    return (unsigned short)(u >> 16);
}

// ---------------------------------------------------------------------------
// Row L2-normalize + convert f32 -> bf16.  One wave per row. Pads extra rows
// (rows..rowsPad) with zeros so the GEMM can read a full final chunk safely.
// ---------------------------------------------------------------------------
__global__ void normalize_bf16(const float* __restrict__ src,
                               unsigned short* __restrict__ dst,
                               int rows, int rowsPad, int D) {
    const int wave = threadIdx.x >> 5;
    const int lane = threadIdx.x & 31;
    const long long row = (long long)blockIdx.x * 8 + wave;
    if (row >= rowsPad) return;

    unsigned short* out = dst + row * (long long)D;
    if (row >= rows) {                       // zero padding rows
        for (int d = lane; d < D; d += 32) out[d] = 0;
        return;
    }
    const float* in = src + row * (long long)D;
    float ss = 0.f;
    for (int d = lane * 4; d < D; d += 128) {
        float4 v = *(const float4*)(in + d);
        ss += v.x * v.x + v.y * v.y + v.z * v.z + v.w * v.w;
    }
    #pragma unroll
    for (int off = 16; off > 0; off >>= 1) ss += __shfl_xor(ss, off, 32);
    const float inv = rsqrtf(ss);
    for (int d = lane * 4; d < D; d += 128) {
        float4 v = *(const float4*)(in + d);
        ushort4 o;
        o.x = f32_to_bf16_rne(v.x * inv);
        o.y = f32_to_bf16_rne(v.y * inv);
        o.z = f32_to_bf16_rne(v.z * inv);
        o.w = f32_to_bf16_rne(v.w * inv);
        *(ushort4*)(out + d) = o;
    }
}

// Build a 16x32 bf16 fragment (A layout; B 32x16 is the mirrored same pattern)
// per ISA: lanes 0-15 hold K {0..7, 16..23}, lanes 16-31 hold K {8..15, 24..31}
// for row = lane & 15.  `rowp` already points at [row, kb]; `half` = lane>>4.
__device__ __forceinline__ v16bf load_frag(const unsigned short* rowp, int half) {
    v8bf g0 = *(const v8bf*)(rowp + half * 8);
    v8bf g1 = *(const v8bf*)(rowp + 16 + half * 8);
    return __builtin_shufflevector(g0, g1, 0, 1, 2, 3, 4, 5, 6, 7,
                                           8, 9, 10, 11, 12, 13, 14, 15);
}

// Pin the interleave: 2 A-loads, then 8x (2 B-loads, 1 WMMA).  Because of the
// ping-pong buffers, each WMMA consumes fragments loaded one k-step earlier,
// so waits should settle well above loadcnt==0.
__device__ __forceinline__ void sched_interleave() {
    __builtin_amdgcn_sched_group_barrier(SGB_VMEMREAD, 2, 0);   // A frag
    #pragma unroll
    for (int j = 0; j < 8; ++j) {
        __builtin_amdgcn_sched_group_barrier(SGB_VMEMREAD, 2, 0); // B frag j
        __builtin_amdgcn_sched_group_barrier(SGB_WMMA, 1, 0);     // WMMA j
    }
}

// ---------------------------------------------------------------------------
// GEMM (64 rows x 512 cols per block, K = D) + per-chunk top-20.
// grid = (B/64, nchunks); blocks with the same chunk launch adjacently so the
// 1 MB bf16 train chunk is served from L2 to all 16 row-blocks.
// K-loop is ping-pong double-buffered (requires D % 64 == 0; D = 1024 here).
// ---------------------------------------------------------------------------
__global__ void __launch_bounds__(256)
knn_gemm_topk(const unsigned short* __restrict__ fq,   // [B, D] bf16
              const unsigned short* __restrict__ tf,   // [Npad, D] bf16
              float* __restrict__ pvals,               // [B, nchunks, 20]
              int*   __restrict__ pidx,                // [B, nchunks, 20]
              int B_, int N, int D, int nchunks) {
    extern __shared__ float lds[];                     // [64][LDS_STRIDE]

    const int tid   = threadIdx.x;
    const int wave  = tid >> 5;
    const int lane  = tid & 31;
    const int half  = lane >> 4;        // K-group selector
    const int mr    = lane & 15;        // row/col within tile
    const int mtile = wave & 3;         // 4 M-tiles of 16 rows
    const int ngrp  = wave >> 2;        // 2 wave groups interleave N-tiles
    const int rowblock   = blockIdx.x;
    const int chunk      = blockIdx.y;
    const long long chunk_base = (long long)chunk * NCHUNK;

    const unsigned short* arow =
        fq + (long long)(rowblock * 64 + mtile * 16 + mr) * D;

    // ---- GEMM: two passes of 8 N-tiles each (keeps accumulators at 64 VGPRs)
    for (int hn = 0; hn < 2; ++hn) {
        const unsigned short* bptr[8];
        #pragma unroll
        for (int j = 0; j < 8; ++j) {
            long long brow = chunk_base + (long long)(hn * 256 + (ngrp + 2 * j) * 16 + mr);
            bptr[j] = tf + brow * D;
        }
        v8f acc[8];
        #pragma unroll
        for (int j = 0; j < 8; ++j)
            acc[j] = (v8f){0.f, 0.f, 0.f, 0.f, 0.f, 0.f, 0.f, 0.f};

        v16bf a0, a1, b0[8], b1[8];

        // prologue: fill buffer 0 with k-step 0
        a0 = load_frag(arow, half);
        #pragma unroll
        for (int j = 0; j < 8; ++j) b0[j] = load_frag(bptr[j], half);

        // steady state: load the other buffer while computing this one
        for (int kb = 0; kb <= D - 128; kb += 64) {
            a1 = load_frag(arow + kb + 32, half);
            #pragma unroll
            for (int j = 0; j < 8; ++j) b1[j] = load_frag(bptr[j] + kb + 32, half);
            #pragma unroll
            for (int j = 0; j < 8; ++j)
                acc[j] = __builtin_amdgcn_wmma_f32_16x16x32_bf16(
                    false, a0, false, b0[j], (short)0, acc[j], false, false);
            sched_interleave();

            a0 = load_frag(arow + kb + 64, half);
            #pragma unroll
            for (int j = 0; j < 8; ++j) b0[j] = load_frag(bptr[j] + kb + 64, half);
            #pragma unroll
            for (int j = 0; j < 8; ++j)
                acc[j] = __builtin_amdgcn_wmma_f32_16x16x32_bf16(
                    false, a1, false, b1[j], (short)0, acc[j], false, false);
            sched_interleave();
        }

        // epilogue: last two k-steps (D-64 in b0, load D-32 into b1)
        a1 = load_frag(arow + D - 32, half);
        #pragma unroll
        for (int j = 0; j < 8; ++j) b1[j] = load_frag(bptr[j] + D - 32, half);
        #pragma unroll
        for (int j = 0; j < 8; ++j)
            acc[j] = __builtin_amdgcn_wmma_f32_16x16x32_bf16(
                false, a0, false, b0[j], (short)0, acc[j], false, false);
        sched_interleave();
        #pragma unroll
        for (int j = 0; j < 8; ++j)
            acc[j] = __builtin_amdgcn_wmma_f32_16x16x32_bf16(
                false, a1, false, b1[j], (short)0, acc[j], false, false);

        // C layout: element i of lane -> M = i + 8*half, N = lane & 15
        const int rbase = mtile * 16 + half * 8;
        #pragma unroll
        for (int j = 0; j < 8; ++j) {
            const int col = hn * 256 + (ngrp + 2 * j) * 16 + mr;
            #pragma unroll
            for (int i = 0; i < 8; ++i)
                lds[(rbase + i) * LDS_STRIDE + col] = acc[j][i];
        }
    }
    __syncthreads();

    // ---- per-chunk top-20: 8 rows per wave, iterative wave-argmax + mask
    long long rem = (long long)N - chunk_base;
    const int valid = (rem >= NCHUNK) ? NCHUNK : (int)rem;  // cols in this chunk

    for (int rr = 0; rr < 8; ++rr) {
        const int r = wave * 8 + rr;
        const int grow = rowblock * 64 + r;
        if (grow >= B_) continue;
        float* srow = lds + r * LDS_STRIDE;
        float* pv = pvals + ((long long)grow * nchunks + chunk) * KTOP;
        int*   pi = pidx  + ((long long)grow * nchunks + chunk) * KTOP;

        for (int t = 0; t < KTOP; ++t) {
            float v = -INFINITY;
            int   p = -1;
            for (int j = lane; j < valid; j += 32) {
                float x = srow[j];
                if (x > v) { v = x; p = j; }
            }
            #pragma unroll
            for (int off = 16; off > 0; off >>= 1) {
                float ov = __shfl_xor(v, off, 32);
                int   op = __shfl_xor(p, off, 32);
                if (ov > v) { v = ov; p = op; }
            }
            if (lane == 0) {
                pv[t] = v;
                pi[t] = (p >= 0) ? (int)(chunk_base + p) : 0;
                if (p >= 0) srow[p] = -INFINITY;   // mask out for next round
            }
            // same-wave LDS ops are in-order (DScnt); no barrier needed
        }
    }
}

// ---------------------------------------------------------------------------
// Merge partial top-20s per query row, softmax, scatter votes.
// One wave per row; extraction order yields descending sims (top-10 = prefix).
// ---------------------------------------------------------------------------
__global__ void __launch_bounds__(32)
knn_merge_scatter(const float* __restrict__ pvals,
                  const int*   __restrict__ pidx,
                  const int*   __restrict__ labels,
                  float* __restrict__ out,       // [2, B, 1000]
                  int B_, int nchunks) {
    extern __shared__ float smem[];
    const int b    = blockIdx.x;
    const int lane = threadIdx.x;
    const int ncand = nchunks * KTOP;

    float* sval = smem;
    int*   sidx = (int*)(smem + ncand);
    __shared__ float topv[KTOP];
    __shared__ int   topl[KTOP];

    const float* pv = pvals + (long long)b * ncand;
    const int*   pi = pidx  + (long long)b * ncand;
    for (int j = lane; j < ncand; j += 32) {
        sval[j] = pv[j];
        sidx[j] = pi[j];
    }
    __syncthreads();

    for (int t = 0; t < KTOP; ++t) {
        float v = -INFINITY;
        int   p = -1;
        for (int j = lane; j < ncand; j += 32) {
            float x = sval[j];
            if (x > v) { v = x; p = j; }
        }
        #pragma unroll
        for (int off = 16; off > 0; off >>= 1) {
            float ov = __shfl_xor(v, off, 32);
            int   op = __shfl_xor(p, off, 32);
            if (ov > v) { v = ov; p = op; }
        }
        if (lane == 0) {
            topv[t] = v;
            topl[t] = (p >= 0) ? labels[sidx[p]] : 0;
            if (p >= 0) sval[p] = -INFINITY;
        }
    }
    __syncthreads();

    float* out0 = out + (long long)b * NUM_CLASSES;                 // k=10
    float* out1 = out + ((long long)B_ + b) * NUM_CLASSES;          // k=20
    for (int c = lane; c < NUM_CLASSES; c += 32) { out0[c] = 0.f; out1[c] = 0.f; }
    __syncthreads();

    if (lane == 0) {
        float e[KTOP];
        const float m = topv[0];
        float s10 = 0.f, s20 = 0.f;
        for (int i = 0; i < KTOP; ++i) {
            e[i] = __expf((topv[i] - m) * TINV);
            if (i < KSMALL) s10 += e[i];
            s20 += e[i];
        }
        const float r10 = 1.f / s10, r20 = 1.f / s20;
        for (int i = 0; i < KSMALL; ++i) out0[topl[i]] += e[i] * r10;
        for (int i = 0; i < KTOP;   ++i) out1[topl[i]] += e[i] * r20;
    }
}

// ---------------------------------------------------------------------------
extern "C" void kernel_launch(void* const* d_in, const int* in_sizes, int n_in,
                              void* d_out, int out_size, void* d_ws, size_t ws_size,
                              hipStream_t stream) {
    const float* fq     = (const float*)d_in[0];   // [B, D]
    const float* tr     = (const float*)d_in[1];   // [N, D]
    const int*   labels = (const int*)d_in[2];     // [N]

    const int N = in_sizes[2];
    const int D = (int)((long long)in_sizes[1] / N);
    const int B = (int)((long long)in_sizes[0] / D);
    const int nchunks = (N + NCHUNK - 1) / NCHUNK;
    const int Npad    = nchunks * NCHUNK;

    // workspace layout (256B aligned slices)
    char* ws = (char*)d_ws;
    size_t off = 0;
    auto alloc = [&](size_t bytes) {
        void* p = ws + off;
        off = (off + bytes + 255) & ~(size_t)255;
        return p;
    };
    unsigned short* tf_bf = (unsigned short*)alloc((size_t)Npad * D * 2);
    unsigned short* fq_bf = (unsigned short*)alloc((size_t)B * D * 2);
    float* pvals = (float*)alloc((size_t)B * nchunks * KTOP * 4);
    int*   pidx  = (int*)  alloc((size_t)B * nchunks * KTOP * 4);

    normalize_bf16<<<(Npad + 7) / 8, 256, 0, stream>>>(tr, tf_bf, N, Npad, D);
    normalize_bf16<<<(B + 7) / 8,    256, 0, stream>>>(fq, fq_bf, B, B, D);

    dim3 grid(B / 64, nchunks);   // x = row-block (fast) -> chunk reuse in L2
    const size_t gemm_lds = (size_t)64 * LDS_STRIDE * sizeof(float);
    knn_gemm_topk<<<grid, 256, gemm_lds, stream>>>(fq_bf, tf_bf, pvals, pidx,
                                                   B, N, D, nchunks);

    const size_t merge_lds = (size_t)nchunks * KTOP * 8;
    knn_merge_scatter<<<B, 32, merge_lds, stream>>>(pvals, pidx, labels,
                                                    (float*)d_out, B, nchunks);
}
